// LSTMClassifier_7962869366963
// MI455X (gfx1250) — compile-verified, matrix operand order checked by reading, not run on hardware
//
#include <hip/hip_runtime.h>
#include <cmath>

typedef _Float16 h16;
typedef __attribute__((ext_vector_type(8)))  _Float16 v8h;
typedef __attribute__((ext_vector_type(16))) _Float16 v16h;
typedef __attribute__((ext_vector_type(8)))  float    v8f;

#define B_   64
#define T_   2048
#define FIN  128
#define H_   256
#define OUT_ 10
#define K_   5
#define CC   64
#define TC_  1022   // (2048-5)/2 + 1

__device__ __forceinline__ v16h frag_split(const h16* p) {
    // two 16-byte chunks: halves [0..7] and [16..23] relative to p
    v8h a = *(const v8h*)(p);
    v8h b = *(const v8h*)(p + 16);
    return __builtin_shufflevector(a, b, 0,1,2,3,4,5,6,7,8,9,10,11,12,13,14,15);
}

__device__ __forceinline__ v16h frag_packed(const h16* p) {
    // 32 contiguous bytes per lane (packed B fragment)
    v8h a = *(const v8h*)(p);
    v8h b = *(const v8h*)(p + 8);
    return __builtin_shufflevector(a, b, 0,1,2,3,4,5,6,7,8,9,10,11,12,13,14,15);
}

__device__ __forceinline__ v8f wmma_f16(v16h a, v16h b, v8f c) {
    return __builtin_amdgcn_wmma_f32_16x16x32_f16(false, a, false, b, (short)0, c, false, false);
}

__device__ __forceinline__ float sigm(float x) { return 1.0f / (1.0f + __expf(-x)); }

// ---------------- Kernel 1: L2 normalize over time axis, emit f16 -------------
__global__ __launch_bounds__(512) void norm_kernel(const float* __restrict__ in,
                                                   h16* __restrict__ xn) {
    __shared__ float part[512];
    __shared__ float invs[FIN];
    int b = blockIdx.x, tid = threadIdx.x;
    int f = tid & 127, s = tid >> 7;           // 4 time-slices per feature
    const float* base = in + (long)b * T_ * FIN + f;
    float acc = 0.0f;
    for (int t = s; t < T_; t += 4) { float v = base[(long)t * FIN]; acc += v * v; }
    part[tid] = acc;
    __syncthreads();
    if (s == 0) {
        float sum = part[f] + part[128 + f] + part[256 + f] + part[384 + f];
        invs[f] = 1.0f / fmaxf(sqrtf(sum), 1e-12f);
    }
    __syncthreads();
    float iv = invs[f];
    h16* ob = xn + (long)b * T_ * FIN + f;
    for (int t = s; t < T_; t += 4) ob[(long)t * FIN] = (h16)(base[(long)t * FIN] * iv);
}

// ---------------- Kernel 2: pack weights into WMMA B-fragment layout ----------
// B-fragment: lane L holds col n = L&15; k_local(e) = (e&7) + 8*((e>>3)*2 + (L>>4))
__global__ __launch_bounds__(256) void pack_kernel(const float* __restrict__ conv_w,
                                                   const float* __restrict__ w_ih,
                                                   const float* __restrict__ w_hh,
                                                   const float* __restrict__ b_ih,
                                                   const float* __restrict__ b_hh,
                                                   h16* __restrict__ convw_pk,
                                                   h16* __restrict__ wih_pk,
                                                   h16* __restrict__ whh_pk,
                                                   float* __restrict__ bsum) {
    const int NCV = 20 * 4 * 512;    // conv: K=640(20kt), N=64(4ct)
    const int NIH = 2 * 64 * 512;    // w_ih: K=64(2kt),  N=1024(64ct)
    const int NHH = 8 * 64 * 512;    // w_hh: K=256(8kt), N=1024(64ct)
    int idx = blockIdx.x * 256 + threadIdx.x;
    if (idx < NCV) {
        int fidx = idx >> 9, off = idx & 511;
        int kt = fidx >> 2, ct = fidx & 3;
        int L = off >> 4, e = off & 15;
        int n = ct * 16 + (L & 15);
        int k = kt * 32 + (e & 7) + 8 * (((e >> 3) << 1) + (L >> 4));
        int f = k & 127, ko = k >> 7;                 // window flat = t_off*128 + f
        convw_pk[idx] = (h16)conv_w[(n * FIN + f) * K_ + ko];
    } else if (idx < NCV + NIH) {
        int j = idx - NCV;
        int fidx = j >> 9, off = j & 511;
        int kt = fidx >> 6, ct = fidx & 63;
        int L = off >> 4, e = off & 15;
        int n = ct * 16 + (L & 15);
        int k = kt * 32 + (e & 7) + 8 * (((e >> 3) << 1) + (L >> 4));
        wih_pk[j] = (h16)w_ih[n * CC + k];
    } else if (idx < NCV + NIH + NHH) {
        int j = idx - NCV - NIH;
        int fidx = j >> 9, off = j & 511;
        int kt = fidx >> 6, ct = fidx & 63;
        int L = off >> 4, e = off & 15;
        int n = ct * 16 + (L & 15);
        int k = kt * 32 + (e & 7) + 8 * (((e >> 3) << 1) + (L >> 4));
        whh_pk[j] = (h16)w_hh[n * H_ + k];
    } else if (idx < NCV + NIH + NHH + 1024) {
        int n = idx - NCV - NIH - NHH;
        bsum[n] = b_ih[n] + b_hh[n];
    }
}

// ---------------- Kernel 3: conv1d as im2col WMMA GEMM + bias + ReLU ----------
// M = B*Tc = 65408 (m = b*Tc+tc), K = 640, N = 64. Output lstm[(tc*64+b)*64+n] f16
__global__ __launch_bounds__(256) void conv_gemm_kernel(const h16* __restrict__ xn,
                                                        const h16* __restrict__ convw_pk,
                                                        const float* __restrict__ conv_b,
                                                        h16* __restrict__ lstm) {
    int wave = (blockIdx.x << 3) + (threadIdx.x >> 5);
    int lane = threadIdx.x & 31;
    int mt = wave >> 2, ct = wave & 3;
    int r = lane & 15, hf = lane >> 4;
    int m = mt * 16 + r;
    int b = m / TC_, tc = m - b * TC_;
    const h16* abase = xn + (((long)(b * T_ + tc * 2)) << 7) + hf * 8;
    const h16* bbase = convw_pk + (size_t)ct * 512 + lane * 16;
    v8f acc = {};
#pragma unroll
    for (int kt = 0; kt < 20; ++kt) {
        v16h a  = frag_split(abase + kt * 32);
        v16h bf = frag_packed(bbase + (size_t)kt * 4 * 512);
        acc = wmma_f16(a, bf, acc);
    }
    int n = ct * 16 + r;
    float bias = conv_b[n];
#pragma unroll
    for (int e = 0; e < 8; ++e) {
        int mm = mt * 16 + e + 8 * hf;          // C layout: M = vgpr + 8*(lane>=16)
        int b2 = mm / TC_, tc2 = mm - b2 * TC_;
        float v = fmaxf(acc[e] + bias, 0.0f);
        lstm[((long)tc2 * B_ + b2) * CC + n] = (h16)v;
    }
}

// ---------------- Kernel 4: persistent single-WGP LSTM scan + final linear ----
// 32 waves. wave = rhalf*16 + jt: rows [rhalf*32, +32), hidden cols [jt*16, +16).
// Gates for i/f/g/o at col-tiles g*16+jt -> all 4 gates in-wave; c-state in VGPRs.
__global__ __launch_bounds__(1024, 1) void lstm_seq_kernel(const h16* __restrict__ lstm,
                                                           const h16* __restrict__ wih_pk,
                                                           const h16* __restrict__ whh_pk,
                                                           const float* __restrict__ bsum,
                                                           const float* __restrict__ lin_w,
                                                           const float* __restrict__ lin_b,
                                                           float* __restrict__ out) {
    __shared__ alignas(16) h16 hbuf[2][B_ * H_];   // 2 x 32 KB, double-buffered h (f16)
    int tid = threadIdx.x;
    int wave = tid >> 5, lane = tid & 31;
    int r = lane & 15, hf = lane >> 4;
    int jt = wave & 15, rhalf = wave >> 4;

    for (int i = tid; i < B_ * H_; i += 1024) hbuf[0][i] = (h16)0.0f;
    __syncthreads();

    v8f cst[2] = {};                                // c-state tiles (2 row-tiles)
    float bz[4];
#pragma unroll
    for (int g = 0; g < 4; ++g) bz[g] = bsum[g * 256 + jt * 16 + r];

    for (int t = 0; t < TC_; ++t) {
        const h16* hread = hbuf[t & 1];
        const h16* xrow  = lstm + (long)t * (B_ * CC);
        __builtin_prefetch(xrow + B_ * CC, 0, 1);   // next step's conv activations

        v8f acc[4][2];
#pragma unroll
        for (int g = 0; g < 4; ++g)
#pragma unroll
            for (int rt2 = 0; rt2 < 2; ++rt2)
#pragma unroll
                for (int e = 0; e < 8; ++e) acc[g][rt2][e] = bz[g];

#pragma unroll
        for (int kt = 0; kt < 10; ++kt) {           // kt 0..1: x*W_ih^T, kt 2..9: h*W_hh^T
            v16h bf[4];
#pragma unroll
            for (int g = 0; g < 4; ++g) {
                const h16* bp;
                if (kt < 2) bp = wih_pk + ((size_t)(kt * 64 + (g * 16 + jt)) * 512) + lane * 16;
                else        bp = whh_pk + ((size_t)((kt - 2) * 64 + (g * 16 + jt)) * 512) + lane * 16;
                bf[g] = frag_packed(bp);
            }
#pragma unroll
            for (int rt2 = 0; rt2 < 2; ++rt2) {
                int rt = (rhalf << 1) + rt2;
                v16h a;
                if (kt < 2) a = frag_split(xrow  + ((rt * 16 + r) << 6) + kt * 32 + hf * 8);
                else        a = frag_split(hread + ((rt * 16 + r) << 8) + (kt - 2) * 32 + hf * 8);
#pragma unroll
                for (int g = 0; g < 4; ++g) acc[g][rt2] = wmma_f16(a, bf[g], acc[g][rt2]);
            }
        }

        h16* hw = hbuf[(t + 1) & 1];
#pragma unroll
        for (int rt2 = 0; rt2 < 2; ++rt2) {
#pragma unroll
            for (int e = 0; e < 8; ++e) {
                float gi = acc[0][rt2][e], gf = acc[1][rt2][e];
                float gg = acc[2][rt2][e], go = acc[3][rt2][e];
                float c = sigm(gf) * cst[rt2][e] + sigm(gi) * tanhf(gg);
                cst[rt2][e] = c;
                float h = sigm(go) * tanhf(c);
                int mm = ((rhalf << 1) + rt2) * 16 + e + 8 * hf;
                hw[mm * H_ + jt * 16 + r] = (h16)h;
            }
        }
        __syncthreads();
    }

    // final linear: h_T @ lin_w^T + lin_b  -> [64,10]; final h is in hbuf[0]
    if (tid < B_ * OUT_) {
        int m = tid / OUT_, o = tid - m * OUT_;
        const h16* hrow = hbuf[0] + m * H_;
        float s = lin_b[o];
        for (int k = 0; k < H_; ++k) s += (float)hrow[k] * lin_w[o * H_ + k];
        out[tid] = s;
    }
}

// -------------------------------- launcher -----------------------------------
extern "C" void kernel_launch(void* const* d_in, const int* in_sizes, int n_in,
                              void* d_out, int out_size, void* d_ws, size_t ws_size,
                              hipStream_t stream) {
    const float* input  = (const float*)d_in[0];
    // d_in[1] = r (unused zeros)
    const float* conv_w = (const float*)d_in[2];
    const float* conv_b = (const float*)d_in[3];
    const float* w_ih   = (const float*)d_in[4];
    const float* w_hh   = (const float*)d_in[5];
    const float* b_ih   = (const float*)d_in[6];
    const float* b_hh   = (const float*)d_in[7];
    const float* lin_w  = (const float*)d_in[8];
    const float* lin_b  = (const float*)d_in[9];
    float* out = (float*)d_out;

    char* ws = (char*)d_ws;
    size_t off = 0;
    h16* xn       = (h16*)(ws + off); off += (size_t)B_ * T_ * FIN * 2;       // 33.55 MB
    h16* lstm     = (h16*)(ws + off); off += (size_t)TC_ * B_ * CC * 2;       //  8.37 MB
    h16* convw_pk = (h16*)(ws + off); off += (size_t)20 * 4 * 512 * 2;
    h16* wih_pk   = (h16*)(ws + off); off += (size_t)2 * 64 * 512 * 2;
    h16* whh_pk   = (h16*)(ws + off); off += (size_t)8 * 64 * 512 * 2;
    float* bsum   = (float*)(ws + off); off += 1024 * 4;

    norm_kernel<<<B_, 512, 0, stream>>>(input, xn);
    pack_kernel<<<1444, 256, 0, stream>>>(conv_w, w_ih, w_hh, b_ih, b_hh,
                                          convw_pk, wih_pk, whh_pk, bsum);
    conv_gemm_kernel<<<2044, 256, 0, stream>>>(xn, convw_pk, conv_b, lstm);
    lstm_seq_kernel<<<1, 1024, 0, stream>>>(lstm, wih_pk, whh_pk, bsum, lin_w, lin_b, out);
}
